// MultiHeadAttention_55482387529795
// MI455X (gfx1250) — matrix operand order repaired and checked by reading, not compiled, not run on hardware
//
#include <hip/hip_runtime.h>
#include <hip/hip_bf16.h>
#include <cstdint>
#include <cstddef>

// ---------------------------------------------------------------------------
// MI455X / gfx1250 multi-head attention, bf16 WMMA path (wave32),
// with async-to-LDS double-buffered K/V staging in the flash kernel.
// ---------------------------------------------------------------------------

typedef __attribute__((ext_vector_type(16))) __bf16 v16bf;
typedef __attribute__((ext_vector_type(8)))  __bf16 v8bf;
typedef __attribute__((ext_vector_type(8)))  float  v8f;

#define S_LEN   2048
#define N_HEADS 16
#define DHEAD   64
#define DMODEL  1024
#define BATCH   4
#define NROW    (BATCH * S_LEN)   // 8192 total rows across batch

// fp32 -> bf16, round to nearest even
static __device__ __forceinline__ __bf16 f2bf(float f) {
  unsigned u = __builtin_bit_cast(unsigned, f);
  unsigned r = u + 0x7FFFu + ((u >> 16) & 1u);
  unsigned short hbits = (unsigned short)(r >> 16);
  return __builtin_bit_cast(__bf16, hbits);
}

static __device__ __forceinline__ v8f wmma_bf16(v16bf a, v16bf b, v8f c) {
  // 8 args: (neg_a, A, neg_b, B, c_mod, C, reuse_a, reuse_b)
  return __builtin_amdgcn_wmma_f32_16x16x32_bf16(
      false, a, false, b, (short)0, c, false, false);
}

// Build a 16-element fragment register from two contiguous 16B runs.
static __device__ __forceinline__ v16bf make_frag(const __bf16* lo,
                                                  const __bf16* hi) {
  v8bf l = *(const v8bf*)lo;
  v8bf h = *(const v8bf*)hi;
  v16bf r;
#pragma unroll
  for (int i = 0; i < 8; ++i) { r[i] = l[i]; r[8 + i] = h[i]; }
  return r;
}

// Async DMA: 16 bytes global -> LDS, tracked by ASYNCcnt (no VGPR data).
// VDST = 32-bit LDS byte offset (low 32 bits of the shared-memory address).
static __device__ __forceinline__ void async_load_b128(const __bf16* gptr,
                                                       __bf16* lptr) {
  asm volatile("global_load_async_to_lds_b128 %0, %1, off"
               :
               : "v"((unsigned)(size_t)lptr),
                 "v"((unsigned long long)(size_t)gptr)
               : "memory");
}

static __device__ __forceinline__ void wait_async0() {
  asm volatile("s_wait_asynccnt 0x0" ::: "memory");
}

// ---------------------------------------------------------------------------
// Kernel 1: fp32 -> bf16 conversion
// ---------------------------------------------------------------------------
__global__ __launch_bounds__(256)
void mha_cvt_bf16_kernel(const float* __restrict__ in, __bf16* __restrict__ out,
                         int n) {
  int i = blockIdx.x * blockDim.x + threadIdx.x;
  if (i < n) out[i] = f2bf(in[i]);
}

// ---------------------------------------------------------------------------
// Kernel 2: C[M,N] = A[M,K] @ W[N,K]^T  (bf16 in, fp32 accum)
// mode 0: write bf16 to per-head layout  out[(b*H+h)][s][d]      (Q, K)
// mode 1: write bf16 to transposed head  out[(b*H+h)][d][s]      (Vt)
// mode 2: write fp32 row-major           out[m][n]               (final)
// Block: 256 threads (8 waves); block tile 128x128; wave tile 32x64.
// ---------------------------------------------------------------------------
__global__ __launch_bounds__(256)
void mha_gemm_wt_kernel(const __bf16* __restrict__ A,
                        const __bf16* __restrict__ W,
                        void* __restrict__ out,
                        int M, int N, int K, int mode) {
  const int lane = threadIdx.x & 31;
  const int wave = threadIdx.x >> 5;
  const int r = lane & 15;
  const int h = lane >> 4;
  const int mbase = blockIdx.y * 128 + (wave >> 1) * 32;
  const int nbase = blockIdx.x * 128 + (wave & 1) * 64;

  v8f acc[2][4];
#pragma unroll
  for (int i = 0; i < 2; ++i)
#pragma unroll
    for (int j = 0; j < 4; ++j)
#pragma unroll
      for (int e = 0; e < 8; ++e) acc[i][j][e] = 0.0f;

  for (int kk = 0; kk < K; kk += 32) {
    v16bf afrag[2];
#pragma unroll
    for (int i = 0; i < 2; ++i) {
      const __bf16* p = A + (size_t)(mbase + i * 16 + r) * K + kk + 8 * h;
      afrag[i] = make_frag(p, p + 16);
    }
    v16bf bfrag[4];
#pragma unroll
    for (int j = 0; j < 4; ++j) {
      const __bf16* p = W + (size_t)(nbase + j * 16 + r) * K + kk + 8 * h;
      bfrag[j] = make_frag(p, p + 16);
    }
#pragma unroll
    for (int i = 0; i < 2; ++i)
#pragma unroll
      for (int j = 0; j < 4; ++j)
        acc[i][j] = wmma_bf16(afrag[i], bfrag[j], acc[i][j]);
  }

#pragma unroll
  for (int i = 0; i < 2; ++i)
#pragma unroll
    for (int j = 0; j < 4; ++j) {
      const int n = nbase + j * 16 + r;
#pragma unroll
      for (int jj = 0; jj < 8; ++jj) {
        const int m = mbase + i * 16 + 8 * h + jj;
        const float v = acc[i][j][jj];
        if (mode == 2) {
          ((float*)out)[(size_t)m * N + n] = v;
        } else {
          const int b = m >> 11, s = m & (S_LEN - 1);
          const int hd = n >> 6, d = n & 63;
          size_t idx;
          if (mode == 0)
            idx = (((size_t)(b * N_HEADS + hd)) * S_LEN + s) * DHEAD + d;
          else
            idx = (((size_t)(b * N_HEADS + hd)) * DHEAD + d) * S_LEN + s;
          ((__bf16*)out)[idx] = f2bf(v);
        }
      }
    }
}

// ---------------------------------------------------------------------------
// Kernel 3: flash attention. One wave owns 16 query rows; the 8 waves of a
// block all belong to the same (b,h), so K/V chunks are staged into LDS once
// per block via GLOBAL_LOAD_ASYNC_TO_LDS_B128 (ASYNCcnt), double-buffered.
// Q,K per-head row-major [S,64]; V transposed per-head [64,S].
// Output z written bf16 into [B*S, DMODEL] (columns h*64+d).
// ---------------------------------------------------------------------------
__global__ __launch_bounds__(256)
void mha_flash_kernel(const __bf16* __restrict__ Q,
                      const __bf16* __restrict__ Kmat,
                      const __bf16* __restrict__ Vt,
                      const unsigned char* __restrict__ mask,
                      __bf16* __restrict__ Zout) {
  __shared__ __attribute__((aligned(16))) __bf16 kbuf[2][32][64];  // 2x4 KiB
  __shared__ __attribute__((aligned(16))) __bf16 vbuf[2][64][32];  // 2x4 KiB
  __shared__ __attribute__((aligned(16))) __bf16 plds[8][16][32];  // 8 KiB

  const int tid  = threadIdx.x;
  const int lane = tid & 31;
  const int wave = tid >> 5;
  const int r = lane & 15;
  const int h = lane >> 4;

  const int tile = blockIdx.x * 8 + wave;   // global 16-row query tile id
  const int bh = blockIdx.x >> 4;           // uniform across block (128/8=16)
  const int q0 = (tile & 127) * 16;
  const int b = bh >> 4;
  const int hd = bh & 15;

  const __bf16* Qh = Q    + (size_t)bh * S_LEN * DHEAD;
  const __bf16* Kh = Kmat + (size_t)bh * S_LEN * DHEAD;
  const __bf16* Vh = Vt   + (size_t)bh * DHEAD * S_LEN;
  const unsigned char* mb = mask + (size_t)b * S_LEN * S_LEN;

  // Stage one 32-key chunk of K (32x64) and Vt (64x32): 2 async b128/thread.
  const int krow = tid >> 3, kseg = tid & 7;   // K: 32 rows x 8 segs
  const int vrow = tid >> 2, vseg = tid & 3;   // V: 64 rows x 4 segs
  auto stage = [&](int buf, int k0) {
    async_load_b128(Kh + (size_t)(k0 + krow) * DHEAD + kseg * 8,
                    &kbuf[buf][krow][kseg * 8]);
    async_load_b128(Vh + (size_t)vrow * S_LEN + k0 + vseg * 8,
                    &vbuf[buf][vrow][vseg * 8]);
  };

  // Q fragments for this wave's 16 rows (K-dim 64 -> 2 fragments), kept live.
  v16bf qf[2];
#pragma unroll
  for (int i = 0; i < 2; ++i) {
    const __bf16* p = Qh + (size_t)(q0 + r) * DHEAD + i * 32 + 8 * h;
    qf[i] = make_frag(p, p + 16);
  }

  float mi[8], li[8];
  v8f zacc[4];
#pragma unroll
  for (int j = 0; j < 8; ++j) { mi[j] = -3.0e38f; li[j] = 0.0f; }
#pragma unroll
  for (int t = 0; t < 4; ++t)
#pragma unroll
    for (int j = 0; j < 8; ++j) zacc[t][j] = 0.0f;

  const float scale = 0.125f;   // 1/sqrt(64)
  const int NCHUNK = S_LEN / 32;

  stage(0, 0);

  for (int it = 0; it < NCHUNK; ++it) {
    const int k0 = it * 32;
    const int buf = it & 1;

    wait_async0();        // own chunk-it loads complete
    __syncthreads();      // everyone's complete; prev buffer fully consumed
    if (it + 1 < NCHUNK) stage(buf ^ 1, k0 + 32);

    // ---- scores: S = Q @ K^T for 32 keys (two 16-col tiles), from LDS ----
    v8f sc[2];
#pragma unroll
    for (int c = 0; c < 2; ++c) {
#pragma unroll
      for (int j = 0; j < 8; ++j) sc[c][j] = 0.0f;
#pragma unroll
      for (int i = 0; i < 2; ++i) {
        const __bf16* p = &kbuf[buf][c * 16 + r][i * 32 + 8 * h];
        v16bf kf = make_frag(p, p + 16);
        sc[c] = wmma_bf16(qf[i], kf, sc[c]);
      }
    }

    // ---- scale + mask ----
#pragma unroll
    for (int c = 0; c < 2; ++c)
#pragma unroll
      for (int j = 0; j < 8; ++j) {
        const int q = q0 + 8 * h + j;
        const unsigned char mk = mb[(size_t)q * S_LEN + k0 + c * 16 + r];
        const float s = sc[c][j] * scale;
        sc[c][j] = mk ? -1.0e9f : s;
      }

    // ---- online softmax (row stats across the 16 lanes of each half) ----
#pragma unroll
    for (int j = 0; j < 8; ++j) {
      float cm = fmaxf(sc[0][j], sc[1][j]);
#pragma unroll
      for (int d = 1; d < 16; d <<= 1) cm = fmaxf(cm, __shfl_xor(cm, d, 32));
      const float nm = fmaxf(mi[j], cm);
      const float corr = __expf(mi[j] - nm);
      const float p0 = __expf(sc[0][j] - nm);
      const float p1 = __expf(sc[1][j] - nm);
      sc[0][j] = p0; sc[1][j] = p1;
      float rs = p0 + p1;
#pragma unroll
      for (int d = 1; d < 16; d <<= 1) rs += __shfl_xor(rs, d, 32);
      li[j] = li[j] * corr + rs;
      mi[j] = nm;
#pragma unroll
      for (int t = 0; t < 4; ++t) zacc[t][j] *= corr;
    }

    // ---- transpose P (C-layout) -> A-fragment layout via per-wave LDS ----
#pragma unroll
    for (int c = 0; c < 2; ++c)
#pragma unroll
      for (int j = 0; j < 8; ++j)
        plds[wave][8 * h + j][c * 16 + r] = f2bf(sc[c][j]);
    asm volatile("s_wait_dscnt 0x0" ::: "memory");  // same-wave DS in order
    v16bf pf = make_frag(&plds[wave][r][8 * h], &plds[wave][r][16 + 8 * h]);
    asm volatile("" ::: "memory");

    // ---- z += P @ V  (B-fragments from LDS-staged transposed V) ----
#pragma unroll
    for (int t = 0; t < 4; ++t) {
      const __bf16* p = &vbuf[buf][t * 16 + r][8 * h];
      v16bf vf = make_frag(p, p + 16);
      zacc[t] = wmma_bf16(pf, vf, zacc[t]);
    }
  }

  // ---- normalize and scatter z (bf16) into [B*S, DMODEL] ----
#pragma unroll
  for (int t = 0; t < 4; ++t)
#pragma unroll
    for (int j = 0; j < 8; ++j) {
      const int q = q0 + 8 * h + j;
      const int n = hd * DHEAD + t * 16 + r;
      const float v = zacc[t][j] / li[j];
      Zout[(size_t)(b * S_LEN + q) * DMODEL + n] = f2bf(v);
    }
}

// ---------------------------------------------------------------------------
// Host launcher
// ---------------------------------------------------------------------------
extern "C" void kernel_launch(void* const* d_in, const int* in_sizes, int n_in,
                              void* d_out, int out_size, void* d_ws,
                              size_t ws_size, hipStream_t stream) {
  (void)in_sizes; (void)n_in; (void)out_size; (void)ws_size;

  const unsigned char* mask = (const unsigned char*)d_in[0];
  const float* x  = (const float*)d_in[1];
  const float* Wq = (const float*)d_in[2];
  const float* Wk = (const float*)d_in[3];
  const float* Wv = (const float*)d_in[4];
  const float* Wo = (const float*)d_in[5];
  float* out = (float*)d_out;

  char* ws = (char*)d_ws;
  __bf16* xb  = (__bf16*)(ws);                          // 16 MiB
  __bf16* wqb = (__bf16*)(ws + (size_t)(16u << 20));    //  2 MiB each
  __bf16* wkb = (__bf16*)(ws + (size_t)(18u << 20));
  __bf16* wvb = (__bf16*)(ws + (size_t)(20u << 20));
  __bf16* wob = (__bf16*)(ws + (size_t)(22u << 20));
  __bf16* Qb  = (__bf16*)(ws + (size_t)(24u << 20));    // 16 MiB
  __bf16* Kb  = (__bf16*)(ws + (size_t)(40u << 20));    // 16 MiB
  __bf16* Vtb = (__bf16*)(ws + (size_t)(56u << 20));    // 16 MiB
  __bf16* Zb  = (__bf16*)(ws + (size_t)(72u << 20));    // 16 MiB -> 88 total

  const int nx = NROW * DMODEL;      // 8,388,608
  const int nw = DMODEL * DMODEL;    // 1,048,576
  mha_cvt_bf16_kernel<<<(nx + 255) / 256, 256, 0, stream>>>(x,  xb,  nx);
  mha_cvt_bf16_kernel<<<(nw + 255) / 256, 256, 0, stream>>>(Wq, wqb, nw);
  mha_cvt_bf16_kernel<<<(nw + 255) / 256, 256, 0, stream>>>(Wk, wkb, nw);
  mha_cvt_bf16_kernel<<<(nw + 255) / 256, 256, 0, stream>>>(Wv, wvb, nw);
  mha_cvt_bf16_kernel<<<(nw + 255) / 256, 256, 0, stream>>>(Wo, wob, nw);

  dim3 gg(DMODEL / 128, NROW / 128);   // (8, 64)
  mha_gemm_wt_kernel<<<gg, 256, 0, stream>>>(xb, wqb, Qb,  NROW, DMODEL, DMODEL, 0);
  mha_gemm_wt_kernel<<<gg, 256, 0, stream>>>(xb, wkb, Kb,  NROW, DMODEL, DMODEL, 0);
  mha_gemm_wt_kernel<<<gg, 256, 0, stream>>>(xb, wvb, Vtb, NROW, DMODEL, DMODEL, 1);

  const int ntiles = BATCH * N_HEADS * (S_LEN / 16);  // 8192 wave tiles
  mha_flash_kernel<<<ntiles / 8, 256, 0, stream>>>(Qb, Kb, Vtb, mask, Zb);

  mha_gemm_wt_kernel<<<gg, 256, 0, stream>>>(Zb, wob, out, NROW, DMODEL, DMODEL, 2);
}
